// ContextBERTSelfAttention1_33938831573051
// MI455X (gfx1250) — matrix operand order repaired and checked by reading, not compiled, not run on hardware
//
#include <hip/hip_runtime.h>
#include <hip/hip_bf16.h>
#include <cstdint>

typedef __attribute__((ext_vector_type(16))) _Float16 v16h;
typedef __attribute__((ext_vector_type(8)))  float    v8f;

union FragH {               // WMMA 16-bit operand fragment: 8 VGPRs, 2 f16 each
    v16h     v;
    uint32_t u[8];
    _Float16 h[16];
};

#define S_LEN  4096
#define E_DIM  768
#define NH     12
#define HD     64
#define BAND_T 513          // 2*W+1
#define BPAD   544          // band columns padded to 17*32

// ---------------------------------------------------------------------------
// Kernel 1: projection GEMM  out(s,n) = X(s,:)@W(:,n) + b(n), stored as f16 in
// [H][S][64] layout. 256 threads = 8 waves, 128x64 tile, K-chunks of 32.
// A staged row-major (K contiguous), B staged TRANSPOSED (Bsh[n][k]) so every
// fragment register is one contiguous 32-bit LDS word.
// ---------------------------------------------------------------------------
__global__ __launch_bounds__(256) void gemm_proj(const float* __restrict__ X,
                                                 const float* __restrict__ Wm,
                                                 const float* __restrict__ bias,
                                                 _Float16* __restrict__ out16)
{
    __shared__ _Float16 Ash[128][40];   // [row][k], padded stride (4B aligned)
    __shared__ _Float16 Bsh[64][40];    // [n][k], transposed stage

    const int tid  = threadIdx.x;
    const int wave = tid >> 5;
    const int lane = tid & 31;
    const int mlane = lane & 15;
    const int g     = lane >> 4;
    const int m0 = blockIdx.x * 128;
    const int n0 = blockIdx.y * 64;

    v8f acc[4] = {};

    for (int k0 = 0; k0 < E_DIM; k0 += 32) {
        // prefetch next K-chunk (gfx1250 global_prefetch path)
        if (k0 + 32 < E_DIM) {
            __builtin_prefetch(&X[(size_t)(m0 + (tid >> 1)) * E_DIM + k0 + 32], 0, 1);
            __builtin_prefetch(&Wm[(size_t)(k0 + 32 + (tid >> 3)) * E_DIM + n0], 0, 1);
        }
        // stage A chunk (128x32), coalesced
        #pragma unroll
        for (int i = 0; i < 16; ++i) {
            int l = i * 256 + tid;
            int r = l >> 5, c = l & 31;
            Ash[r][c] = (_Float16)X[(size_t)(m0 + r) * E_DIM + k0 + c];
        }
        // stage B chunk (32x64) transposed into Bsh[n][k], coalesced reads
        #pragma unroll
        for (int i = 0; i < 8; ++i) {
            int l = i * 256 + tid;
            int r = l >> 6, c = l & 63;        // r = k, c = n
            Bsh[c][r] = (_Float16)Wm[(size_t)(k0 + r) * E_DIM + n0 + c];
        }
        __syncthreads();

        // A fragment: lane m = lane%16; K pairs contiguous -> b32 LDS reads
        FragH a;
        {
            const uint32_t* arow = (const uint32_t*)&Ash[wave * 16 + mlane][0];
            #pragma unroll
            for (int v = 0; v < 8; ++v) {
                int kb = ((v >> 2) << 4) + ((v & 3) << 1) + (g << 3);
                a.u[v] = arow[kb >> 1];
            }
        }
        #pragma unroll
        for (int nt = 0; nt < 4; ++nt) {
            FragH b;
            const uint32_t* brow = (const uint32_t*)&Bsh[nt * 16 + mlane][0];
            #pragma unroll
            for (int v = 0; v < 8; ++v) {
                int kb = ((v >> 2) << 4) + ((v & 3) << 1) + (g << 3);
                b.u[v] = brow[kb >> 1];
            }
            acc[nt] = __builtin_amdgcn_wmma_f32_16x16x32_f16(
                false, a.v, false, b.v, (short)0, acc[nt], false, false);
        }
        __syncthreads();
    }

    // write out: C/D layout M = v + 8*g, N = lane%16
    #pragma unroll
    for (int nt = 0; nt < 4; ++nt) {
        #pragma unroll
        for (int v = 0; v < 8; ++v) {
            int m    = v + (g << 3);
            int srow = m0 + wave * 16 + m;
            int col  = n0 + nt * 16 + mlane;
            float val = acc[nt][v] + bias[col];
            int h = col >> 6, d = col & 63;
            out16[((size_t)h * S_LEN + srow) * HD + d] = (_Float16)val;
        }
    }
}

// ---------------------------------------------------------------------------
// Kernel 2: fused banded QK (attn + quasi), masking, softmax, sigmoid,
// lambda-gating -> new_probs. One block = 16 queries x 1 head, 128 thr/4 waves.
// Band scores live only in LDS (never hit HBM). Fragment loads are packed b32.
// ---------------------------------------------------------------------------
__global__ __launch_bounds__(128) void band_attn(
    const _Float16* __restrict__ q16,  const _Float16* __restrict__ k16,
    const _Float16* __restrict__ cq16, const _Float16* __restrict__ ck16,
    const int* __restrict__ amask,
    const float* __restrict__ Wlqc, const float* __restrict__ blqc,
    const float* __restrict__ Wlqq, const float* __restrict__ blqq,
    const float* __restrict__ Wlkc, const float* __restrict__ blkc,
    const float* __restrict__ Wlkk, const float* __restrict__ blkk,
    float* __restrict__ new_probs)
{
    __shared__ float    attn_s[16][BPAD];  // masked attn scores
    __shared__ _Float16 qsig_s[16][BPAD];  // sigmoid(masked quasi)
    __shared__ float    red[16][8];
    __shared__ float    lam_s[16];

    const int tid   = threadIdx.x;
    const int wave  = tid >> 5;
    const int lane  = tid & 31;
    const int mlane = lane & 15;
    const int g     = lane >> 4;
    const int s0 = blockIdx.x * 16;
    const int h  = blockIdx.y;
    const float scale = 0.125f;            // 1/sqrt(64)

    // A fragments: queries (unscaled q) and quasi-queries (unscaled cq),
    // two K-chunks of 32 dims; K pairs are memory-contiguous -> b32 loads.
    FragH aq[2], ac[2];
    {
        const int srow = s0 + mlane;
        const uint32_t* qrow  = (const uint32_t*)(q16  + ((size_t)h * S_LEN + srow) * HD);
        const uint32_t* cqrow = (const uint32_t*)(cq16 + ((size_t)h * S_LEN + srow) * HD);
        #pragma unroll
        for (int kc = 0; kc < 2; ++kc) {
            #pragma unroll
            for (int v = 0; v < 8; ++v) {
                int d = kc * 32 + ((v >> 2) << 4) + ((v & 3) << 1) + (g << 3);
                aq[kc].u[v] = qrow[d >> 1];
                ac[kc].u[v] = cqrow[d >> 1];
            }
        }
    }

    // pass 0: quasi = cq.cq * scale^2 -> sigmoid ; pass 1: attn = q.k * scale
    for (int pass = 0; pass < 2; ++pass) {
        const _Float16* kbase = pass ? k16 : cq16;
        const float sc = pass ? scale : scale * scale;
        for (int kt = wave; kt < 33; kt += 4) {
            int key = s0 - 256 + kt * 16 + mlane;     // B-matrix lane = key
            bool kvalid = (key >= 0) && (key < S_LEN);
            const uint32_t* krow =
                (const uint32_t*)(kbase + ((size_t)h * S_LEN + (kvalid ? key : 0)) * HD);
            // prefetch next key tile for this wave
            {
                int nkey = key + 64;
                if (nkey >= 0 && nkey < S_LEN)
                    __builtin_prefetch(kbase + ((size_t)h * S_LEN + nkey) * HD, 0, 1);
            }
            v8f acc = {};
            #pragma unroll
            for (int kc = 0; kc < 2; ++kc) {
                FragH b;
                #pragma unroll
                for (int v = 0; v < 8; ++v) {
                    int kd = kc * 32 + ((v >> 2) << 4) + ((v & 3) << 1) + (g << 3);
                    b.u[v] = kvalid ? krow[kd >> 1] : 0u;
                }
                const FragH& A = pass ? aq[kc] : ac[kc];
                acc = __builtin_amdgcn_wmma_f32_16x16x32_f16(
                    false, A.v, false, b.v, (short)0, acc, false, false);
            }
            // scatter to LDS with position/attention masking
            #pragma unroll
            for (int v = 0; v < 8; ++v) {
                int m = v + (g << 3);
                int c = kt * 16 + mlane;
                int kg = s0 - 256 + c;
                bool valid = (kg >= 0) && (kg < S_LEN);
                float addm = (valid && amask[kg] != 0) ? -10000.0f : 0.0f;
                float val = valid ? acc[v] * sc + addm : -1.0e9f;
                if (pass) attn_s[m][c] = val;
                else      qsig_s[m][c] = (_Float16)(1.0f / (1.0f + __expf(-val)));
            }
        }
    }
    __syncthreads();

    // per-row lambda = 1 - (sigmoid(scale*cq.Wlqc + q.Wlqq + b) + sigmoid(ck.Wlkc + k.Wlkk + b))
    if (tid < 16) {
        int s = s0 + tid;
        float dq = 0.f, dcq = 0.f, dk = 0.f, dck = 0.f;
        for (int d = 0; d < HD; ++d) {
            size_t base = ((size_t)h * S_LEN + s) * HD + d;
            dcq += (float)cq16[base] * Wlqc[d];
            dq  += (float)q16[base]  * Wlqq[d];
            dck += (float)ck16[base] * Wlkc[d];
            dk  += (float)k16[base]  * Wlkk[d];
        }
        float lq = 1.0f / (1.0f + __expf(-(scale * dcq + blqc[0] + dq + blqq[0])));
        float lk = 1.0f / (1.0f + __expf(-(dck + blkc[0] + dk + blkk[0])));
        lam_s[tid] = 1.0f - (lq + lk);
    }
    __syncthreads();

    // softmax over the 513-wide band + combine; 8 threads per query row
    const int row = tid >> 3, sub = tid & 7;
    float mx = -1.0e30f;
    for (int t = sub; t < BAND_T; t += 8) mx = fmaxf(mx, attn_s[row][row + t]);
    red[row][sub] = mx;
    __syncthreads();
    mx = red[row][0];
    #pragma unroll
    for (int i = 1; i < 8; ++i) mx = fmaxf(mx, red[row][i]);
    float sum = 0.f;
    for (int t = sub; t < BAND_T; t += 8) sum += __expf(attn_s[row][row + t] - mx);
    __syncthreads();
    red[row][sub] = sum;
    __syncthreads();
    sum = 0.f;
    #pragma unroll
    for (int i = 0; i < 8; ++i) sum += red[row][i];
    const float inv = 1.0f / sum;

    const int s = s0 + row;
    const bool idxm = amask[s] != 0;
    const float lam = lam_s[row];
    float* np = new_probs + ((size_t)s * NH + h) * BAND_T;
    for (int t = sub; t < BAND_T; t += 8) {
        float p  = __expf(attn_s[row][row + t] - mx) * inv;
        float nv = p + lam * (float)qsig_s[row][row + t];
        np[t] = idxm ? 0.0f : nv;
    }
}

// ---------------------------------------------------------------------------
// Kernel 3: banded PV. out(s, h*64+d) = sum_t new_probs(s,h,t) * V(s+t-256,h,d)
// One wave per (16-query block, head). 17 K-chunks of 32 band columns.
// ---------------------------------------------------------------------------
__global__ __launch_bounds__(32) void band_pv(const float* __restrict__ new_probs,
                                              const _Float16* __restrict__ v16,
                                              float* __restrict__ outp)
{
    const int lane  = threadIdx.x & 31;
    const int mlane = lane & 15;
    const int g     = lane >> 4;
    const int s0 = blockIdx.x * 16;
    const int h  = blockIdx.y;

    v8f acc[4] = {};

    for (int ckk = 0; ckk < 17; ++ckk) {
        const int c0 = ckk * 32;
        // A fragment: shifted probability matrix p[j][c] = np[j][c-j], f32->f16
        FragH a;
        const int j = mlane;
        const float* nprow = new_probs + ((size_t)(s0 + j) * NH + h) * BAND_T;
        #pragma unroll
        for (int v = 0; v < 8; ++v) {
            int cb = c0 + ((v >> 2) << 4) + ((v & 3) << 1) + (g << 3);
            #pragma unroll
            for (int e = 0; e < 2; ++e) {
                int c  = cb + e;
                int t  = c - j;
                int kg = s0 - 256 + c;
                float pv = (t >= 0 && t < BAND_T && kg >= 0 && kg < S_LEN)
                               ? nprow[t] : 0.0f;
                a.h[2 * v + e] = (_Float16)pv;
            }
        }
        #pragma unroll
        for (int nt = 0; nt < 4; ++nt) {
            FragH b;
            #pragma unroll
            for (int v = 0; v < 8; ++v) {
                #pragma unroll
                for (int e = 0; e < 2; ++e) {
                    int c  = c0 + ((v >> 2) << 4) + ((v & 3) << 1) + (g << 3) + e;   // key column
                    int kg = s0 - 256 + c;
                    _Float16 val = (_Float16)0.0f;
                    if (kg >= 0 && kg < S_LEN)
                        val = v16[((size_t)h * S_LEN + kg) * HD + nt * 16 + mlane];
                    b.h[2 * v + e] = val;
                }
            }
            acc[nt] = __builtin_amdgcn_wmma_f32_16x16x32_f16(
                false, a.v, false, b.v, (short)0, acc[nt], false, false);
        }
    }

    #pragma unroll
    for (int nt = 0; nt < 4; ++nt) {
        #pragma unroll
        for (int v = 0; v < 8; ++v) {
            int m    = v + (g << 3);
            int srow = s0 + m;
            int col  = h * 64 + nt * 16 + mlane;
            outp[(size_t)srow * E_DIM + col] = acc[nt][v];
        }
    }
}

// ---------------------------------------------------------------------------
extern "C" void kernel_launch(void* const* d_in, const int* in_sizes, int n_in,
                              void* d_out, int out_size, void* d_ws, size_t ws_size,
                              hipStream_t stream)
{
    const float* hidden  = (const float*)d_in[0];
    const float* context = (const float*)d_in[1];
    const int*   amask   = (const int*)d_in[2];
    // d_in[3] (is_index_masked) is derived from amask != 0; recompute on device.
    const float* Wq  = (const float*)d_in[4];  const float* bq  = (const float*)d_in[5];
    const float* Wk  = (const float*)d_in[6];  const float* bk  = (const float*)d_in[7];
    const float* Wv  = (const float*)d_in[8];  const float* bv  = (const float*)d_in[9];
    const float* Wcq = (const float*)d_in[10]; const float* bcq = (const float*)d_in[11];
    const float* Wck = (const float*)d_in[12]; const float* bck = (const float*)d_in[13];
    const float* Wlqc = (const float*)d_in[14]; const float* blqc = (const float*)d_in[15];
    const float* Wlqq = (const float*)d_in[16]; const float* blqq = (const float*)d_in[17];
    const float* Wlkc = (const float*)d_in[18]; const float* blkc = (const float*)d_in[19];
    const float* Wlkk = (const float*)d_in[20]; const float* blkk = (const float*)d_in[21];

    // workspace: 5 f16 projection tensors in [H][S][64] layout (~30 MB)
    const size_t proj_elems = (size_t)NH * S_LEN * HD;
    _Float16* q16  = (_Float16*)d_ws;
    _Float16* k16  = q16  + proj_elems;
    _Float16* v16p = k16  + proj_elems;
    _Float16* cq16 = v16p + proj_elems;
    _Float16* ck16 = cq16 + proj_elems;

    float* outp      = (float*)d_out;                       // (S, E)
    float* new_probs = outp + (size_t)S_LEN * E_DIM;        // (S, H, 513)

    dim3 gg(S_LEN / 128, E_DIM / 64);
    gemm_proj<<<gg, 256, 0, stream>>>(hidden,  Wq,  bq,  q16);
    gemm_proj<<<gg, 256, 0, stream>>>(hidden,  Wk,  bk,  k16);
    gemm_proj<<<gg, 256, 0, stream>>>(hidden,  Wv,  bv,  v16p);
    gemm_proj<<<gg, 256, 0, stream>>>(context, Wcq, bcq, cq16);
    gemm_proj<<<gg, 256, 0, stream>>>(context, Wck, bck, ck16);

    dim3 gb(S_LEN / 16, NH);
    band_attn<<<gb, 128, 0, stream>>>(q16, k16, cq16, ck16, amask,
                                      Wlqc, blqc, Wlqq, blqq,
                                      Wlkc, blkc, Wlkk, blkk, new_probs);
    band_pv<<<gb, 32, 0, stream>>>(new_probs, v16p, outp);
}
